// Encoder_3272765080210
// MI455X (gfx1250) — matrix operand order repaired and checked by reading, not compiled, not run on hardware
//
#include <hip/hip_runtime.h>
#include <hip/hip_bf16.h>

// ---------------------------------------------------------------------------
// BiGRU encoder for MI455X (gfx1250): all matmuls on v_wmma_f32_16x16x32_bf16
// (bf16 operands, f32 accumulate). Serial recurrence -> 2 kernels per step.
// Each wave computes a 32x64 output tile (2 M-frags x 4 N-frags, B reused).
// ---------------------------------------------------------------------------

typedef __bf16 bf16;
typedef __attribute__((ext_vector_type(16))) __bf16 bf16x16;
typedef __attribute__((ext_vector_type(8)))  float  f32x8;

constexpr int kB = 256;   // batch
constexpr int kT = 200;   // time
constexpr int kD = 256;   // input dim
constexpr int kH = 512;   // hidden
constexpr int kG3 = 3 * kH;       // 1536
constexpr float kClip = 5.0f;
constexpr float kVarMin = 1e-4f;

__device__ __forceinline__ bf16 f2bf(float f) {
  unsigned u = __builtin_bit_cast(unsigned, f);
  u += 0x7FFFu + ((u >> 16) & 1u);          // round-to-nearest-even
  unsigned short h = (unsigned short)(u >> 16);
  return __builtin_bit_cast(bf16, h);
}

__device__ __forceinline__ f32x8 zero8() {
  f32x8 z = {0.f, 0.f, 0.f, 0.f, 0.f, 0.f, 0.f, 0.f};
  return z;
}

// A fragment: 16x32 (MxK) bf16, row-major source, row stride lda elements.
// Lanes 0-15: M=lane, K = kb+{0..15}; lanes 16-31: M=lane-16,
// K = kb+{16..31}. Two 16B loads per lane.
__device__ __forceinline__ bf16x16 load_a_frag(const bf16* __restrict__ base,
                                               int lda, int kb) {
  const int lane = threadIdx.x & 31;
  const int row  = lane & 15;
  const int kh   = (lane >> 4) << 4;        // 0 or 16
  const bf16* p = base + (size_t)row * lda + kb + kh;
  union { bf16x16 v; uint4 u[2]; } t;
  t.u[0] = *(const uint4*)(p);
  t.u[1] = *(const uint4*)(p + 8);
  return t.v;
}

// B fragment: 32x16 (KxN) bf16 built from weight matrix W[n,k] row-major
// (so B[k][n] = W[n][k]).  Lane holds column n = lane&15; lanes 0-15 K=kb..kb+15,
// lanes 16-31 K=kb+16..kb+31, packed 2/VGPR ascending. 16 contiguous bf16/lane.
__device__ __forceinline__ bf16x16 load_b_frag(const bf16* __restrict__ wrow0,
                                               int ldb, int kb) {
  const int lane = threadIdx.x & 31;
  const int n  = lane & 15;
  const int k0 = kb + ((lane >> 4) << 4);   // 0 or 16
  const bf16* p = wrow0 + (size_t)n * ldb + k0;
  union { bf16x16 v; uint4 u[2]; } t;
  t.u[0] = *(const uint4*)(p);
  t.u[1] = *(const uint4*)(p + 8);
  return t.v;
}

__device__ __forceinline__ f32x8 wmma_bf16(bf16x16 a, bf16x16 b, f32x8 c) {
  return __builtin_amdgcn_wmma_f32_16x16x32_bf16(false, a, false, b,
                                                 (short)0, c, false, false);
}

// 32x64 register-blocked GEMM accumulation: C[2][4] += A(32xK) * B(64xK)^T.
// aBase: row mbase of A (stride lda); wBase: row nbase of W (stride ldw).
__device__ __forceinline__ void gemm_32x64(f32x8 (&acc)[2][4],
                                           const bf16* __restrict__ aBase, int lda,
                                           const bf16* __restrict__ wBase, int ldw,
                                           int ktot) {
  for (int kb = 0; kb < ktot; kb += 32) {
    bf16x16 a0 = load_a_frag(aBase, lda, kb);
    bf16x16 a1 = load_a_frag(aBase + (size_t)16 * lda, lda, kb);
#pragma unroll
    for (int j = 0; j < 4; ++j) {
      bf16x16 b = load_b_frag(wBase + (size_t)(16 * j) * ldw, ldw, kb);
      acc[0][j] = wmma_bf16(a0, b, acc[0][j]);
      acc[1][j] = wmma_bf16(a1, b, acc[1][j]);
    }
  }
}

// ---------------------------------------------------------------------------
// One-time conversion f32 -> bf16
__global__ __launch_bounds__(256) void cvt_f32_bf16(const float* __restrict__ src,
                                                    bf16* __restrict__ dst, int n) {
  int i = blockIdx.x * 256 + threadIdx.x;
  if (i < n) dst[i] = f2bf(src[i]);
}

// Init hidden state from h0 (broadcast over batch) and zero lag-boundary ci slots.
__global__ __launch_bounds__(256) void init_state(const float* __restrict__ ic_h0,
                                                  const float* __restrict__ ci_h0,
                                                  float* __restrict__ h_ws,
                                                  bf16* __restrict__ hbf,
                                                  float* __restrict__ out_ci) {
  int i = blockIdx.x * 256 + threadIdx.x;
  const int nh = 4 * kB * kH;
  if (i < nh) {
    int g = i / (kB * kH);
    int j = i % kH;
    float v = ((g < 2) ? ic_h0 : ci_h0)[(g & 1) * kH + j];
    h_ws[i] = v;
    hbf[i] = f2bf(v);
  } else {
    int k = i - nh;
    if (k < kB * kH) {            // ci[:, 0, 0:512] = 0   (fwd lag pad)
      int b = k / kH, c = k % kH;
      out_ci[(size_t)b * kT * (2 * kH) + c] = 0.f;
    } else if (k < 2 * kB * kH) { // ci[:, T-1, 512:1024] = 0 (bwd lag pad)
      k -= kB * kH;
      int b = k / kH, c = k % kH;
      out_ci[(size_t)b * kT * (2 * kH) + (size_t)(kT - 1) * (2 * kH) + kH + c] = 0.f;
    }
  }
}

// ---------------------------------------------------------------------------
// Step kernel A: gates z, r  ( [x_t | h] @ [Wih_zr;Whh_zr]^T )  and  xn ( x_t @ Wih_n^T ).
// 192 wave tiles (128 zr + 64 xn) of 32x64 -> 24 blocks of 8 waves; y = 4 dirs.
__global__ __launch_bounds__(256) void gru_step_a(
    const bf16* __restrict__ xbf,      // [B,T,D]
    const bf16* __restrict__ wihbf,    // [4,1536,256]
    const bf16* __restrict__ whhbf,    // [4,1536,512]
    const bf16* __restrict__ hbf,      // [4,B,H]
    const float* __restrict__ h_ws,    // [4,B,H]
    const float* __restrict__ ic_bih, const float* __restrict__ ic_bhh,
    const float* __restrict__ ci_bih, const float* __restrict__ ci_bhh,
    float* __restrict__ z_ws,          // [4,B,H]
    float* __restrict__ xn_ws,         // [4,B,H]
    bf16* __restrict__ rhbf,           // [4,B,H]
    int t) {
  const int g = blockIdx.y;
  const int tile = blockIdx.x * 8 + (threadIdx.x >> 5);
  const int tx = (g & 1) ? (kT - 1 - t) : t;   // backward dirs consume flipped time

  const float* bih = ((g < 2) ? ic_bih : ci_bih) + (g & 1) * kG3;
  const float* bhh = ((g < 2) ? ic_bhh : ci_bhh) + (g & 1) * kG3;
  const bf16* wih = wihbf + (size_t)g * kG3 * kD;
  const bf16* whh = whhbf + (size_t)g * kG3 * kH;
  const bf16* hb  = hbf  + (size_t)g * kB * kH;
  const float* hw = h_ws + (size_t)g * kB * kH;

  const int lane  = threadIdx.x & 31;
  const int nlane = lane & 15;
  const int mo    = (lane >> 4) * 8;

  f32x8 acc[2][4];
#pragma unroll
  for (int mi = 0; mi < 2; ++mi)
#pragma unroll
    for (int j = 0; j < 4; ++j) acc[mi][j] = zero8();

  if (tile < 128) {
    // ---- z|r region: M=256, N=1024, K = 256 (x) + 512 (h) ----
    const int mbase = (tile >> 4) * 32;
    const int nbase = (tile & 15) * 64;
    gemm_32x64(acc, xbf + (size_t)mbase * (kT * kD) + (size_t)tx * kD, kT * kD,
               wih + (size_t)nbase * kD, kD, kD);
    gemm_32x64(acc, hb + (size_t)mbase * kH, kH,
               whh + (size_t)nbase * kH, kH, kH);
#pragma unroll
    for (int j = 0; j < 4; ++j) {
      const int ncol = nbase + 16 * j + nlane;
      const float bsum = bih[ncol] + bhh[ncol];
#pragma unroll
      for (int mi = 0; mi < 2; ++mi) {
#pragma unroll
        for (int v = 0; v < 8; ++v) {
          const int m = mbase + 16 * mi + mo + v;
          const float s = 1.f / (1.f + __expf(-(acc[mi][j][v] + bsum)));
          if (ncol < kH) {                       // z gate
            z_ws[(size_t)g * kB * kH + (size_t)m * kH + ncol] = s;
          } else {                               // r gate -> stash r*h (bf16)
            const int c = ncol - kH;
            const float rh = s * hw[(size_t)m * kH + c];
            rhbf[(size_t)g * kB * kH + (size_t)m * kH + c] = f2bf(rh);
          }
        }
      }
    }
  } else {
    // ---- xn region: M=256, N=512, K=256 ----
    const int tt = tile - 128;
    const int mbase = (tt >> 3) * 32;
    const int nbase = (tt & 7) * 64;
    gemm_32x64(acc, xbf + (size_t)mbase * (kT * kD) + (size_t)tx * kD, kT * kD,
               wih + (size_t)(2 * kH + nbase) * kD, kD, kD);
#pragma unroll
    for (int j = 0; j < 4; ++j) {
      const int ncol = nbase + 16 * j + nlane;
      const float bn = bih[2 * kH + ncol];
#pragma unroll
      for (int mi = 0; mi < 2; ++mi) {
#pragma unroll
        for (int v = 0; v < 8; ++v) {
          const int m = mbase + 16 * mi + mo + v;
          xn_ws[(size_t)g * kB * kH + (size_t)m * kH + ncol] = acc[mi][j][v] + bn;
        }
      }
    }
  }
}

// ---------------------------------------------------------------------------
// Step kernel B: n = tanh(xn + (r*h) @ Whh_n^T + b_n); h' = clip(z*h+(1-z)*n).
// 64 wave tiles of 32x64 over M=256 x N=512 -> 8 blocks of 8 waves; y = 4 dirs.
__global__ __launch_bounds__(256) void gru_step_b(
    const bf16* __restrict__ rhbf,     // [4,B,H]
    const bf16* __restrict__ whhbf,    // [4,1536,512]
    const float* __restrict__ xn_ws,   // [4,B,H]
    const float* __restrict__ z_ws,    // [4,B,H]
    const float* __restrict__ ic_bhh, const float* __restrict__ ci_bhh,
    float* __restrict__ h_ws,          // [4,B,H]
    bf16* __restrict__ hbf,            // [4,B,H]
    float* __restrict__ out_ci,        // d_out + 2*B*ICD, layout [B,T,2H]
    int t) {
  const int g = blockIdx.y;
  const int tile = blockIdx.x * 8 + (threadIdx.x >> 5);
  const int mbase = (tile >> 3) * 32;
  const int nbase = (tile & 7) * 64;

  f32x8 acc[2][4];
#pragma unroll
  for (int mi = 0; mi < 2; ++mi)
#pragma unroll
    for (int j = 0; j < 4; ++j) acc[mi][j] = zero8();

  gemm_32x64(acc, rhbf + (size_t)g * kB * kH + (size_t)mbase * kH, kH,
             whhbf + (size_t)g * kG3 * kH + (size_t)(2 * kH + nbase) * kH, kH, kH);

  const float* bhh = (((g < 2) ? ic_bhh : ci_bhh) + (g & 1) * kG3) + 2 * kH;
  const int lane  = threadIdx.x & 31;
  const int nlane = lane & 15;
  const int mo    = (lane >> 4) * 8;

#pragma unroll
  for (int j = 0; j < 4; ++j) {
    const int ncol = nbase + 16 * j + nlane;
    const float bn = bhh[ncol];
#pragma unroll
    for (int mi = 0; mi < 2; ++mi) {
#pragma unroll
      for (int v = 0; v < 8; ++v) {
        const int m = mbase + 16 * mi + mo + v;
        const size_t idx = (size_t)g * kB * kH + (size_t)m * kH + ncol;
        const float nv = tanhf(acc[mi][j][v] + bn + xn_ws[idx]);
        const float z  = z_ws[idx];
        float hn = z * h_ws[idx] + (1.f - z) * nv;
        hn = fminf(kClip, fmaxf(-kClip, hn));
        h_ws[idx] = hn;
        hbf[idx]  = f2bf(hn);
        if (g == 2) {                      // ci forward: lands at time t+1
          if (t < kT - 1)
            out_ci[(size_t)m * kT * (2 * kH) + (size_t)(t + 1) * (2 * kH) + ncol] = hn;
        } else if (g == 3) {               // ci backward: orig time T-1-t -> slot -1
          const int to = kT - 1 - t;
          if (to >= 1)
            out_ci[(size_t)m * kT * (2 * kH) + (size_t)(to - 1) * (2 * kH) + kH + ncol] = hn;
        }
      }
    }
  }
}

// ---------------------------------------------------------------------------
// Head: ic_params = [hn_f|hn_b] @ lin_W^T + lin_b ; mean + sqrt(exp(logvar)+eps).
// 32 wave tiles of 32x64 (M=256 x N=256) -> 4 blocks of 8 waves.
__global__ __launch_bounds__(256) void head_kernel(
    const bf16* __restrict__ hbf,      // [4,B,H]; g=0 fwd, g=1 bwd
    const bf16* __restrict__ linWbf,   // [256,1024]
    const float* __restrict__ lin_b,   // [256]
    float* __restrict__ out) {
  const int tile = blockIdx.x * 8 + (threadIdx.x >> 5);
  const int mbase = (tile >> 2) * 32;
  const int nbase = (tile & 3) * 64;

  f32x8 acc[2][4];
#pragma unroll
  for (int mi = 0; mi < 2; ++mi)
#pragma unroll
    for (int j = 0; j < 4; ++j) acc[mi][j] = zero8();

  // fwd half (K 0..511) then bwd half (K 512..1023)
  gemm_32x64(acc, hbf + (size_t)mbase * kH, kH,
             linWbf + (size_t)nbase * (2 * kH), 2 * kH, kH);
  gemm_32x64(acc, hbf + (size_t)kB * kH + (size_t)mbase * kH, kH,
             linWbf + (size_t)nbase * (2 * kH) + kH, 2 * kH, kH);

  const int lane  = threadIdx.x & 31;
  const int nlane = lane & 15;
  const int mo    = (lane >> 4) * 8;
#pragma unroll
  for (int j = 0; j < 4; ++j) {
    const int jc = nbase + 16 * j + nlane;
    const float bj = lin_b[jc];
#pragma unroll
    for (int mi = 0; mi < 2; ++mi) {
#pragma unroll
      for (int v = 0; v < 8; ++v) {
        const int m = mbase + 16 * mi + mo + v;
        const float val = acc[mi][j][v] + bj;
        if (jc < 128) {
          out[(size_t)m * 128 + jc] = val;                       // ic_mean
        } else {
          out[32768 + (size_t)m * 128 + (jc - 128)] = sqrtf(__expf(val) + kVarMin);
        }
      }
    }
  }
}

// ---------------------------------------------------------------------------
extern "C" void kernel_launch(void* const* d_in, const int* in_sizes, int n_in,
                              void* d_out, int out_size, void* d_ws, size_t ws_size,
                              hipStream_t stream) {
  const float* data   = (const float*)d_in[0];
  const float* ic_h0  = (const float*)d_in[1];
  const float* ci_h0  = (const float*)d_in[2];
  const float* ic_Wih = (const float*)d_in[3];
  const float* ic_Whh = (const float*)d_in[4];
  const float* ic_bih = (const float*)d_in[5];
  const float* ic_bhh = (const float*)d_in[6];
  const float* ci_Wih = (const float*)d_in[7];
  const float* ci_Whh = (const float*)d_in[8];
  const float* ci_bih = (const float*)d_in[9];
  const float* ci_bhh = (const float*)d_in[10];
  const float* lin_W  = (const float*)d_in[11];
  const float* lin_b  = (const float*)d_in[12];

  float* out    = (float*)d_out;
  float* out_ci = out + 2 * kB * 128;   // after ic_mean, ic_std

  // ---- workspace carving (256B aligned) ----
  char* ws = (char*)d_ws;
  size_t off = 0;
  auto carve = [&](size_t bytes) -> char* {
    char* p = ws + off;
    off = (off + bytes + 255) & ~(size_t)255;
    return p;
  };
  bf16*  xbf    = (bf16*)carve((size_t)kB * kT * kD * sizeof(bf16));
  bf16*  wihbf  = (bf16*)carve((size_t)4 * kG3 * kD * sizeof(bf16));
  bf16*  whhbf  = (bf16*)carve((size_t)4 * kG3 * kH * sizeof(bf16));
  bf16*  linWbf = (bf16*)carve((size_t)256 * 2 * kH * sizeof(bf16));
  float* h_ws   = (float*)carve((size_t)4 * kB * kH * sizeof(float));
  bf16*  hbf    = (bf16*)carve((size_t)4 * kB * kH * sizeof(bf16));
  float* z_ws   = (float*)carve((size_t)4 * kB * kH * sizeof(float));
  float* xn_ws  = (float*)carve((size_t)4 * kB * kH * sizeof(float));
  bf16*  rhbf   = (bf16*)carve((size_t)4 * kB * kH * sizeof(bf16));

  // ---- one-time conversions ----
  int n;
  n = kB * kT * kD;
  cvt_f32_bf16<<<(n + 255) / 256, 256, 0, stream>>>(data, xbf, n);
  n = 2 * kG3 * kD;
  cvt_f32_bf16<<<(n + 255) / 256, 256, 0, stream>>>(ic_Wih, wihbf, n);
  cvt_f32_bf16<<<(n + 255) / 256, 256, 0, stream>>>(ci_Wih, wihbf + (size_t)2 * kG3 * kD, n);
  n = 2 * kG3 * kH;
  cvt_f32_bf16<<<(n + 255) / 256, 256, 0, stream>>>(ic_Whh, whhbf, n);
  cvt_f32_bf16<<<(n + 255) / 256, 256, 0, stream>>>(ci_Whh, whhbf + (size_t)2 * kG3 * kH, n);
  n = 256 * 2 * kH;
  cvt_f32_bf16<<<(n + 255) / 256, 256, 0, stream>>>(lin_W, linWbf, n);

  // ---- init hidden state + zero lag-boundary ci slots ----
  {
    int total = 4 * kB * kH + 2 * kB * kH;   // 786432
    init_state<<<(total + 255) / 256, 256, 0, stream>>>(ic_h0, ci_h0, h_ws, hbf, out_ci);
  }

  // ---- recurrence: 200 steps, all 4 directions per launch ----
  for (int t = 0; t < kT; ++t) {
    gru_step_a<<<dim3(24, 4), 256, 0, stream>>>(
        xbf, wihbf, whhbf, hbf, h_ws,
        ic_bih, ic_bhh, ci_bih, ci_bhh,
        z_ws, xn_ws, rhbf, t);
    gru_step_b<<<dim3(8, 4), 256, 0, stream>>>(
        rhbf, whhbf, xn_ws, z_ws, ic_bhh, ci_bhh,
        h_ws, hbf, out_ci, t);
  }

  // ---- linear head on ic final hidden states ----
  head_kernel<<<4, 256, 0, stream>>>(hbf, linWbf, lin_b, out);

  (void)in_sizes; (void)n_in; (void)out_size; (void)ws_size;
}
